// SelectOffloadKV_4999341932654
// MI455X (gfx1250) — matrix operand order repaired and checked by reading, not compiled, not run on hardware
//
#include <hip/hip_runtime.h>
#include <hip/hip_bf16.h>
#include <math.h>

// ---------------------------------------------------------------------------
// SelectOffloadKV for MI455X (gfx1250, wave32, WMMA)
//
// Roofline: gathers move ~300MB fp32 -> ~13us @ 23.3TB/s (bound).
// Logits GEMM is only 17.2 GFLOP -> made free with v_wmma_f32_16x16x32_f16.
// ---------------------------------------------------------------------------

typedef __attribute__((ext_vector_type(16))) _Float16 v16h;
typedef __attribute__((ext_vector_type(8)))  float    v8f;

#define PRE   64
#define NIMG  4096
#define SUF   512
#define SEQ   (PRE + NIMG + SUF)      // 4672
#define NHEAD 32
#define DHEAD 128
#define TOPK  1024
#define NREM  (NIMG - TOPK)           // 3072
#define QTILE 16
#define KTILE 16
#define NQT   (SUF / QTILE)           // 32 q-tiles per head
#define NKT   (NIMG / KTILE)          // 256 k-tiles
#define SCALE 0.08838834764831845f    // 1/sqrt(128)

// ---------------------------------------------------------------------------
// Kernel Z: zero the score accumulator (ws is not re-poisoned between replays)
// ---------------------------------------------------------------------------
__global__ void zero_scores_kernel(float* __restrict__ p, int n) {
    int i = blockIdx.x * blockDim.x + threadIdx.x;
    if (i < n) p[i] = 0.0f;
}

// ---------------------------------------------------------------------------
// Kernel A: attn_scores[h][k] = sum_q softmax_k(Q_text[h,q,:].K_img[h,k,:]*s)
// One wave (32 lanes) per (head, 16-row q-tile). WMMA 16x16x32 f16->f32.
//
// A-operand layout (16-bit A 16x32, ISA 7.12.2): lane L (g=L>>4, m=L&15)
// holds row m; element e of v16h maps to K = 16g + e (this code's order, same as B)
// B-operand (32x16): lane L (g, n=L&15) holds column n, K = 16g + e.
// C/D (16x16 f32): lane L (g, n) VGPR r -> element [M = 8g + r][N = n].
// ---------------------------------------------------------------------------
__global__ __launch_bounds__(32)
void attn_scores_kernel(const float* __restrict__ q,
                        const float* __restrict__ k,
                        float* __restrict__ scores) {
    const int wg   = blockIdx.x;          // h * NQT + qt
    const int h    = wg / NQT;
    const int qt   = wg % NQT;
    const int lane = threadIdx.x;
    const int g    = lane >> 4;
    const int ln   = lane & 15;

    // ---- load this wave's Q tile (A operands for all 4 K-steps), once ----
    const int qrow = PRE + NIMG + qt * QTILE + ln;     // text query row (seq idx)
    const float* qp = q + ((size_t)h * SEQ + qrow) * DHEAD;

    v16h a[4];
#pragma unroll
    for (int s = 0; s < 4; ++s) {
        const int d0 = 32 * s + 16 * g;
        v16h t;
#pragma unroll
        for (int e = 0; e < 8; ++e)  t[e] = (_Float16)qp[d0 + e];
#pragma unroll
        for (int e = 8; e < 16; ++e) t[e] = (_Float16)qp[d0 + e];
        a[s] = t;
    }

    float M[8], L[8];
#pragma unroll
    for (int r = 0; r < 8; ++r) { M[r] = -1e30f; L[r] = 0.0f; }

    const float* kimg = k + ((size_t)h * SEQ + PRE) * DHEAD;   // image keys

    // ---- pass 1: online per-row (max, sumexp) over all 4096 columns ----
    for (int t = 0; t < NKT; ++t) {
        const float* kp = kimg + (size_t)(t * KTILE + ln) * DHEAD;
        v8f c = {};
#pragma unroll
        for (int s = 0; s < 4; ++s) {
            const float4* kp4 = (const float4*)(kp + 32 * s + 16 * g);
            float4 f0 = kp4[0], f1 = kp4[1], f2 = kp4[2], f3 = kp4[3];
            v16h b;
            b[0]=(_Float16)f0.x; b[1]=(_Float16)f0.y; b[2]=(_Float16)f0.z; b[3]=(_Float16)f0.w;
            b[4]=(_Float16)f1.x; b[5]=(_Float16)f1.y; b[6]=(_Float16)f1.z; b[7]=(_Float16)f1.w;
            b[8]=(_Float16)f2.x; b[9]=(_Float16)f2.y; b[10]=(_Float16)f2.z; b[11]=(_Float16)f2.w;
            b[12]=(_Float16)f3.x; b[13]=(_Float16)f3.y; b[14]=(_Float16)f3.z; b[15]=(_Float16)f3.w;
            c = __builtin_amdgcn_wmma_f32_16x16x32_f16(false, a[s], false, b,
                                                       (short)0, c, false, false);
        }
#pragma unroll
        for (int r = 0; r < 8; ++r) {
            float x  = c[r] * SCALE;
            float nm = fmaxf(M[r], x);
            L[r] = L[r] * __expf(M[r] - nm) + __expf(x - nm);
            M[r] = nm;
        }
    }

    // combine (M,L) across the 16 lanes holding each half of the row block
#pragma unroll
    for (int mask = 1; mask < 16; mask <<= 1) {
#pragma unroll
        for (int r = 0; r < 8; ++r) {
            float oM = __shfl_xor(M[r], mask, 32);
            float oL = __shfl_xor(L[r], mask, 32);
            float nm = fmaxf(M[r], oM);
            L[r] = L[r] * __expf(M[r] - nm) + oL * __expf(oM - nm);
            M[r] = nm;
        }
    }

    // ---- pass 2: recompute tiles, accumulate normalized column sums ----
    for (int t = 0; t < NKT; ++t) {
        const float* kp = kimg + (size_t)(t * KTILE + ln) * DHEAD;
        v8f c = {};
#pragma unroll
        for (int s = 0; s < 4; ++s) {
            const float4* kp4 = (const float4*)(kp + 32 * s + 16 * g);
            float4 f0 = kp4[0], f1 = kp4[1], f2 = kp4[2], f3 = kp4[3];
            v16h b;
            b[0]=(_Float16)f0.x; b[1]=(_Float16)f0.y; b[2]=(_Float16)f0.z; b[3]=(_Float16)f0.w;
            b[4]=(_Float16)f1.x; b[5]=(_Float16)f1.y; b[6]=(_Float16)f1.z; b[7]=(_Float16)f1.w;
            b[8]=(_Float16)f2.x; b[9]=(_Float16)f2.y; b[10]=(_Float16)f2.z; b[11]=(_Float16)f2.w;
            b[12]=(_Float16)f3.x; b[13]=(_Float16)f3.y; b[14]=(_Float16)f3.z; b[15]=(_Float16)f3.w;
            c = __builtin_amdgcn_wmma_f32_16x16x32_f16(false, a[s], false, b,
                                                       (short)0, c, false, false);
        }
        float colsum = 0.0f;
#pragma unroll
        for (int r = 0; r < 8; ++r) {
            float x = c[r] * SCALE;
            colsum += __expf(x - M[r]) / L[r];
        }
        colsum += __shfl_xor(colsum, 16, 32);           // join both row halves
        if (g == 0)
            atomicAdd(scores + h * NIMG + t * KTILE + ln, colsum);
    }
}

// ---------------------------------------------------------------------------
// Kernel B: exact top-k via rank counting (jax.lax.top_k tie semantics:
// descending score, lower index wins).  rank(i) = #{j : s_j > s_i  or
// (s_j == s_i and j < i)}.  rank < TOPK -> selected, output slot = rank.
// Remainder = unselected indices ascending = i - (#selected j < i).
// ---------------------------------------------------------------------------
__global__ __launch_bounds__(512)
void topk_kernel(const float* __restrict__ scores,
                 int* __restrict__ gpu_order,
                 int* __restrict__ rem_order) {
    __shared__ float         s[NIMG];
    __shared__ unsigned char flag[NIMG];
    const int h   = blockIdx.x;
    const int tid = threadIdx.x;

    for (int i = tid; i < NIMG; i += 512) s[i] = scores[h * NIMG + i];
    __syncthreads();

    float si[8];
    int   rank[8];
#pragma unroll
    for (int u = 0; u < 8; ++u) { si[u] = s[tid * 8 + u]; rank[u] = 0; }

    for (int j = 0; j < NIMG; ++j) {
        float sj = s[j];
#pragma unroll
        for (int u = 0; u < 8; ++u) {
            int i = tid * 8 + u;
            rank[u] += (sj > si[u]) || (sj == si[u] && j < i);
        }
    }

#pragma unroll
    for (int u = 0; u < 8; ++u) {
        int i = tid * 8 + u;
        if (rank[u] < TOPK) {
            flag[i] = 1;
            gpu_order[h * TOPK + rank[u]] = i;
        } else {
            flag[i] = 0;
        }
    }
    __syncthreads();

#pragma unroll
    for (int u = 0; u < 8; ++u) {
        int i = tid * 8 + u;
        if (!flag[i]) {
            int cnt = 0;
            for (int j = 0; j < i; ++j) cnt += flag[j];
            rem_order[h * NREM + (i - cnt)] = i;
        }
    }
}

// ---------------------------------------------------------------------------
// Kernel C: gpu K/V = concat(k[:, :64], k[:, 4160:], top_image) per head.
// float4 coalesced copies; this is the bandwidth-bound part of the problem.
// ---------------------------------------------------------------------------
__global__ void gather_gpu_kernel(const float* __restrict__ k,
                                  const float* __restrict__ v,
                                  const int* __restrict__ gpu_order,
                                  float* __restrict__ out) {
    const long total = 2L * NHEAD * 1600 * (DHEAD / 4);
    long tid = (long)blockIdx.x * blockDim.x + threadIdx.x;
    if (tid >= total) return;
    const int c   = (int)(tid & 31);                  // float4 column
    const int row = (int)((tid >> 5) % 1600);
    const int h   = (int)(((tid >> 5) / 1600) % NHEAD);
    const int kv  = (int)(tid / (32L * 1600 * NHEAD));

    int seq;
    if (row < PRE)             seq = row;                              // pre text
    else if (row < PRE + SUF)  seq = NIMG + row;                       // suf text: 4160+(row-64)
    else                       seq = PRE + gpu_order[h * TOPK + (row - (PRE + SUF))];

    const float* src = (kv ? v : k) + ((size_t)h * SEQ + seq) * DHEAD + c * 4;
    float*       dst = out + (size_t)kv * ((size_t)NHEAD * 1600 * DHEAD)
                           + ((size_t)(h * 1600 + row)) * DHEAD + c * 4;
    *(float4*)dst = *(const float4*)src;
}

// ---------------------------------------------------------------------------
// Kernel D: cpu K/V = unselected image rows in ascending original order.
// ---------------------------------------------------------------------------
__global__ void gather_cpu_kernel(const float* __restrict__ k,
                                  const float* __restrict__ v,
                                  const int* __restrict__ rem_order,
                                  float* __restrict__ out) {
    const long total = 2L * NHEAD * NREM * (DHEAD / 4);
    long tid = (long)blockIdx.x * blockDim.x + threadIdx.x;
    if (tid >= total) return;
    const int c   = (int)(tid & 31);
    const int row = (int)((tid >> 5) % NREM);
    const int h   = (int)(((tid >> 5) / NREM) % NHEAD);
    const int kv  = (int)(tid / (32L * NREM * NHEAD));

    const int seq = PRE + rem_order[h * NREM + row];

    const float* src = (kv ? v : k) + ((size_t)h * SEQ + seq) * DHEAD + c * 4;
    float*       dst = out + 2L * NHEAD * 1600 * DHEAD                 // past gpuK+gpuV
                           + (size_t)kv * ((size_t)NHEAD * NREM * DHEAD)
                           + ((size_t)(h * NREM + row)) * DHEAD + c * 4;
    *(float4*)dst = *(const float4*)src;
}

// ---------------------------------------------------------------------------
extern "C" void kernel_launch(void* const* d_in, const int* in_sizes, int n_in,
                              void* d_out, int out_size, void* d_ws, size_t ws_size,
                              hipStream_t stream) {
    const float* q = (const float*)d_in[0];   // (1,32,4672,128) f32
    const float* k = (const float*)d_in[1];
    const float* v = (const float*)d_in[2];
    // d_in[3] image_mask: implied by constants, unused.
    float* out = (float*)d_out;

    // workspace layout (1 MB total)
    float* scores    = (float*)d_ws;                                       // 32*4096 f32
    int*   gpu_order = (int*)((char*)d_ws + (size_t)NHEAD * NIMG * 4);     // 32*1024 i32
    int*   rem_order = (int*)((char*)gpu_order + (size_t)NHEAD * TOPK * 4);// 32*3072 i32

    zero_scores_kernel<<<(NHEAD * NIMG + 255) / 256, 256, 0, stream>>>(scores, NHEAD * NIMG);

    attn_scores_kernel<<<NHEAD * NQT, 32, 0, stream>>>(q, k, scores);

    topk_kernel<<<NHEAD, 512, 0, stream>>>(scores, gpu_order, rem_order);

    {
        const long total = 2L * NHEAD * 1600 * (DHEAD / 4);
        gather_gpu_kernel<<<(int)((total + 255) / 256), 256, 0, stream>>>(k, v, gpu_order, out);
    }
    {
        const long total = 2L * NHEAD * NREM * (DHEAD / 4);
        gather_cpu_kernel<<<(int)((total + 255) / 256), 256, 0, stream>>>(k, v, rem_order, out);
    }
}
